// ActorCriticGRU_40381282517598
// MI455X (gfx1250) — matrix operand order repaired and checked by reading, not compiled, not run on hardware
//
#include <hip/hip_runtime.h>

// ---------- problem constants ----------
#define BATCH 32768
#define IDIM  128
#define HDIM  256
#define ADIM  32
#define BM    64            // batch rows per block
#define K0    384           // layer0 concat K = I + H
#define K1    512           // layer1 concat K = H + H
#define SP0   392           // stride (halves) of bufA0 rows: 384 + 8 pad  (196 dw % 64 = 4)
#define SP1   520           // stride (halves) of bufA1 rows: 512 + 8 pad  (260 dw % 64 = 4)
#define LDS_BYTES ((BM * SP0 + BM * SP1) * 2)   // 116,736

// bf16 weight workspace layout (element offsets in d_ws)
// WC0: 768 rows x 384  -> row n = [ w_ih0[n,0:128] | w_hh0[n,0:256] ]
// WC1: 768 rows x 512  -> row n = [ w_ih1[n,0:256] | w_hh1[n,0:256] ]
// WPV: 48  rows x 256  -> rows 0..31 = w_p, row 32 = w_v, rest 0
#define WC0_OFF 0
#define WC1_OFF 294912
#define WPV_OFF 688128
#define W_TOTAL 700416

// d_out layout: logits [B*32] | value [B] | h_new [2*B*256]
#define OUT_VALUE (BATCH * ADIM)
#define OUT_H     (BATCH * ADIM + BATCH)

typedef __bf16 bf16_t;
typedef bf16_t v16bf __attribute__((ext_vector_type(16)));
typedef bf16_t v8bf  __attribute__((ext_vector_type(8)));
typedef float  v8f   __attribute__((ext_vector_type(8)));

union AB16 { v16bf v; v8bf h[2]; };

__device__ __forceinline__ unsigned short f2bf(float f) {
    unsigned int u = __float_as_uint(f);
    return (unsigned short)((u + 0x7FFFu + ((u >> 16) & 1u)) >> 16);   // RNE
}
__device__ __forceinline__ float bf2f(unsigned short h) {
    return __uint_as_float(((unsigned int)h) << 16);
}
__device__ __forceinline__ float fast_sigmoid(float x) {
    return __builtin_amdgcn_rcpf(1.0f + __expf(-x));
}
__device__ __forceinline__ float fast_tanh(float x) {
    float e = __expf(2.0f * x);
    return (e - 1.0f) * __builtin_amdgcn_rcpf(e + 1.0f);
}

// A fragment (16x32 MxK, bf16) from an LDS tile. ISA layout:
// lanes 0-15 : row M=lane,    K = kb+{0..7, 16..23}
// lanes 16-31: row M=lane-16, K = kb+{8..15, 24..31}
__device__ __forceinline__ v16bf a_frag(const unsigned short* buf, int stride,
                                        int rbase, int kbase, int lane) {
    int r  = rbase + (lane & 15);
    int k0 = kbase + ((lane >> 4) << 3);
    AB16 u;
    u.h[0] = *(const v8bf*)(buf + r * stride + k0);
    u.h[1] = *(const v8bf*)(buf + r * stride + k0 + 16);
    return u.v;
}

// B fragment (32x16 KxN, bf16) from row-major [N_total, K_total] weights.
// lanes 0-15 : N=lane,    K = kb+{0..7, 16..23}
// lanes 16-31: N=lane-16, K = kb+{8..15, 24..31}
__device__ __forceinline__ v16bf b_frag(const unsigned short* w, int ktot,
                                        int nbase, int kbase, int lane) {
    int n  = nbase + (lane & 15);
    int k0 = kbase + ((lane >> 4) << 3);
    const unsigned short* p = w + n * ktot + k0;
    AB16 u;
    u.h[0] = *(const v8bf*)(p);
    u.h[1] = *(const v8bf*)(p + 16);
    return u.v;
}

__device__ __forceinline__ v8f wmma_bf16(v16bf a, v16bf b, v8f c) {
    return __builtin_amdgcn_wmma_f32_16x16x32_bf16(
        false, a, false, b, (short)0, c, false, false);
}

// Fused GRU layer on concatenated-K inputs.
// aBuf: LDS bf16 A tile [BM x KTOT], row stride SPA; cols [XK..KTOT) hold h_old.
// wc:   combined bf16 weights [768 x KTOT]: rows 0..255 r, 256..511 z, 512..767 n.
// Each wave computes all 4 M-tiles for its 2 H-column tiles -> one B fragment
// feeds 4 wmmas (4x less L2 weight traffic).
template <int KTOT, int XK, int SPA, int SPN>
__device__ __forceinline__ void gru_layer_c(
    const unsigned short* aBuf, const unsigned short* wc,
    const float* b_ih, const float* b_hh,
    unsigned short* hNew, float* outH,
    int bm0, int wave, int lane)
{
    for (int jj = 0; jj < 2; ++jj) {
        int jb = wave * 32 + jj * 16;        // H-column base of this 16-col tile
        v8f ar[4] = {}, az[4] = {}, ain[4] = {}, ahn[4] = {};
        for (int kb = 0; kb < KTOT; kb += 32) {
            v16bf br = b_frag(wc, KTOT,       jb, kb, lane);
            v16bf bz = b_frag(wc, KTOT, 256 + jb, kb, lane);
            v16bf bn = b_frag(wc, KTOT, 512 + jb, kb, lane);
            if (kb < XK) {
                #pragma unroll
                for (int mt = 0; mt < 4; ++mt) {
                    v16bf a = a_frag(aBuf, SPA, mt * 16, kb, lane);
                    ar[mt]  = wmma_bf16(a, br, ar[mt]);
                    az[mt]  = wmma_bf16(a, bz, az[mt]);
                    ain[mt] = wmma_bf16(a, bn, ain[mt]);   // i_n part (x/h0_new columns)
                }
            } else {
                #pragma unroll
                for (int mt = 0; mt < 4; ++mt) {
                    v16bf a = a_frag(aBuf, SPA, mt * 16, kb, lane);
                    ar[mt]  = wmma_bf16(a, br, ar[mt]);
                    az[mt]  = wmma_bf16(a, bz, az[mt]);
                    ahn[mt] = wmma_bf16(a, bn, ahn[mt]);   // h_n part (h_old columns)
                }
            }
        }
        // gating: C/D layout -> lane gives N, vgpr v gives M = mt*16 + 8*(lane>>4) + v
        int col = jb + (lane & 15);
        float br_ = b_ih[col]       + b_hh[col];          // r/z biases fold
        float bz_ = b_ih[256 + col] + b_hh[256 + col];
        float bin = b_ih[512 + col];
        float bhn = b_hh[512 + col];
        #pragma unroll
        for (int mt = 0; mt < 4; ++mt) {
            int rsub = mt * 16 + ((lane >> 4) << 3);
            #pragma unroll
            for (int v = 0; v < 8; ++v) {
                int r = rsub + v;
                float hv = bf2f(aBuf[r * SPA + XK + col]);   // h_old lives in A tile
                float rg = fast_sigmoid(ar[mt][v] + br_);
                float zg = fast_sigmoid(az[mt][v] + bz_);
                float ng = fast_tanh(ain[mt][v] + bin + rg * (ahn[mt][v] + bhn));
                float hnew = (1.0f - zg) * ng + zg * hv;
                hNew[r * SPN + col] = f2bf(hnew);
                outH[(bm0 + r) * HDIM + col] = hnew;
            }
        }
    }
}

__global__ void __launch_bounds__(256)
convert_weights_kernel(const float* __restrict__ w_ih0, const float* __restrict__ w_hh0,
                       const float* __restrict__ w_ih1, const float* __restrict__ w_hh1,
                       const float* __restrict__ w_p,   const float* __restrict__ w_v,
                       unsigned short* __restrict__ wb)
{
    int i = blockIdx.x * 256 + threadIdx.x;
    if (i >= W_TOTAL) return;
    float v;
    if (i < WC1_OFF) {                           // layer0: [w_ih0 | w_hh0] rows of 384
        int row = i / K0, c = i % K0;
        v = (c < IDIM) ? w_ih0[row * IDIM + c] : w_hh0[row * HDIM + (c - IDIM)];
    } else if (i < WPV_OFF) {                    // layer1: [w_ih1 | w_hh1] rows of 512
        int r2 = i - WC1_OFF;
        int row = r2 / K1, c = r2 % K1;
        v = (c < HDIM) ? w_ih1[row * HDIM + c] : w_hh1[row * HDIM + (c - HDIM)];
    } else {                                     // [w_p ; w_v ; zeros] rows of 256
        int r = i - WPV_OFF;
        int row = r >> 8, k = r & 255;
        v = (row < ADIM) ? w_p[row * HDIM + k] : ((row == ADIM) ? w_v[k] : 0.0f);
    }
    wb[i] = f2bf(v);
}

__global__ void __launch_bounds__(256)
gru_fused_kernel(const float* __restrict__ x,  const float* __restrict__ h,
                 const float* __restrict__ b_ih0, const float* __restrict__ b_hh0,
                 const float* __restrict__ b_ih1, const float* __restrict__ b_hh1,
                 const float* __restrict__ b_p,   const float* __restrict__ b_v,
                 const unsigned short* __restrict__ wb, float* __restrict__ out)
{
    extern __shared__ unsigned short smem[];
    unsigned short* bufA0 = smem;               // [BM x SP0] : [x | h0_old] -> later h1_new (cols 0..255)
    unsigned short* bufA1 = smem + BM * SP0;    // [BM x SP1] : [h0_new | h1_old]

    int tid  = threadIdx.x;
    int wave = tid >> 5;
    int lane = tid & 31;
    int bm0  = blockIdx.x * BM;

    // ---- stage: global fp32 -> LDS bf16 (each element read exactly once) ----
    for (int i = tid; i < BM * IDIM; i += 256) {            // x -> bufA0 cols 0..127
        int r = i >> 7, c = i & 127;
        bufA0[r * SP0 + c] = f2bf(x[(bm0 + r) * IDIM + c]);
    }
    for (int i = tid; i < BM * HDIM; i += 256) {
        int r = i >> 8, c = i & 255;
        bufA0[r * SP0 + IDIM + c] = f2bf(h[(bm0 + r) * HDIM + c]);                  // h0_old
        bufA1[r * SP1 + HDIM + c] = f2bf(h[BATCH * HDIM + (bm0 + r) * HDIM + c]);   // h1_old
    }
    __syncthreads();

    float* outH0 = out + OUT_H;
    float* outH1 = out + OUT_H + BATCH * HDIM;

    // ---- layer 0: h0_new = GRU([x|h0_old]); h0_new -> bufA1 cols 0..255 ----
    gru_layer_c<K0, IDIM, SP0, SP1>(bufA0, wb + WC0_OFF, b_ih0, b_hh0,
                                    bufA1, outH0, bm0, wave, lane);
    __syncthreads();

    // ---- layer 1: h1_new = GRU([h0_new|h1_old]); h1_new -> dead bufA0 ----
    gru_layer_c<K1, HDIM, SP1, SP0>(bufA1, wb + WC1_OFF, b_ih1, b_hh1,
                                    bufA0, outH1, bm0, wave, lane);
    __syncthreads();

    // ---- heads: [32 logits | value] = h1_new @ w_pv^T, N padded to 48 ----
    // 12 tiles (4 M x 3 N) over 8 waves
    for (int t = wave; t < 12; t += 8) {
        int m = t / 3, jt = t % 3;
        int rbase = m * 16, nb = jt * 16;
        v8f acc{};
        #pragma unroll
        for (int kb = 0; kb < HDIM; kb += 32) {
            v16bf a = a_frag(bufA0, SP0, rbase, kb, lane);
            acc = wmma_bf16(a, b_frag(wb + WPV_OFF, HDIM, nb, kb, lane), acc);
        }
        int n = nb + (lane & 15);
        float bias = (n < ADIM) ? b_p[n] : ((n == ADIM) ? b_v[0] : 0.0f);
        int rsub = rbase + ((lane >> 4) << 3);
        #pragma unroll
        for (int v = 0; v < 8; ++v) {
            float val  = acc[v] + bias;
            int   grow = bm0 + rsub + v;
            if (n < ADIM)       out[grow * ADIM + n]  = val;   // logits
            else if (n == ADIM) out[OUT_VALUE + grow] = val;   // value
        }
    }
}

extern "C" void kernel_launch(void* const* d_in, const int* in_sizes, int n_in,
                              void* d_out, int out_size, void* d_ws, size_t ws_size,
                              hipStream_t stream) {
    const float* x     = (const float*)d_in[0];
    const float* h     = (const float*)d_in[1];
    const float* w_ih0 = (const float*)d_in[2];
    const float* w_hh0 = (const float*)d_in[3];
    const float* b_ih0 = (const float*)d_in[4];
    const float* b_hh0 = (const float*)d_in[5];
    const float* w_ih1 = (const float*)d_in[6];
    const float* w_hh1 = (const float*)d_in[7];
    const float* b_ih1 = (const float*)d_in[8];
    const float* b_hh1 = (const float*)d_in[9];
    const float* w_p   = (const float*)d_in[10];
    const float* b_p   = (const float*)d_in[11];
    const float* w_v   = (const float*)d_in[12];
    const float* b_v   = (const float*)d_in[13];

    unsigned short* wbp = (unsigned short*)d_ws;
    float* out = (float*)d_out;

    (void)hipFuncSetAttribute((const void*)gru_fused_kernel,
                              hipFuncAttributeMaxDynamicSharedMemorySize, LDS_BYTES);

    convert_weights_kernel<<<(W_TOTAL + 255) / 256, 256, 0, stream>>>(
        w_ih0, w_hh0, w_ih1, w_hh1, w_p, w_v, wbp);

    gru_fused_kernel<<<BATCH / BM, 256, LDS_BYTES, stream>>>(
        x, h, b_ih0, b_hh0, b_ih1, b_hh1, b_p, b_v, wbp, out);
}
